// QuantLinear_72189810311631
// MI455X (gfx1250) — matrix operand order repaired and checked by reading, not compiled, not run on hardware
//
#include <hip/hip_runtime.h>

typedef __attribute__((ext_vector_type(16))) _Float16 v16h;
typedef __attribute__((ext_vector_type(8)))  _Float16 v8h;
typedef __attribute__((ext_vector_type(8)))  float    v8f;

constexpr int IN_F       = 4096;
constexpr int OUT_F      = 11008;
constexpr int GROUP      = 128;
constexpr int NUM_GROUPS = IN_F / GROUP;   // 32
constexpr int M_TOTAL    = 4 * 2048;       // 8192

constexpr int BM = 128, BN = 256, BK = 128;   // BK == GROUP: one scale per row per k-tile
constexpr int LSTR = BK + 8;                  // 136 halfs = 272 B row stride (16B aligned)

static __device__ __forceinline__ v16h cat8(v8h lo, v8h hi) {
    v16h r;
#pragma unroll
    for (int i = 0; i < 8; ++i) { r[i] = lo[i]; r[i + 8] = hi[i]; }
    return r;
}

__global__ __launch_bounds__(256)
void QuantLinear_72189810311631_kernel(const float* __restrict__ x,
                                       const int* __restrict__ wq,
                                       const _Float16* __restrict__ scales,
                                       const float* __restrict__ bias,
                                       float* __restrict__ out)
{
    __shared__ _Float16 lA[BM * LSTR];   // x tile, f16, [m][k]         (34.8 KB)
    __shared__ _Float16 lB[BN * LSTR];   // dequant W tile, f16, [n][k] (69.6 KB)

    const int tid  = threadIdx.x;        // 256 threads = 8 wave32
    const int lane = tid & 31;
    const int wid  = tid >> 5;           // 0..7
    const int wm   = wid & 1;            // wave M position: 2 x 64 rows = 128
    const int wn   = wid >> 1;           // wave N position: 4 x 64 cols = 256

    // grid: x = M tiles (fastest) so a fixed 256-wide weight strip is reused
    // while x (134 MB < 192 MB L2) stays L2-resident across the sweep.
    const int mBase = blockIdx.x * BM;
    const int nBase = blockIdx.y * BN;

    // staging decomposition: 256 threads -> 8 rows x 32 (16B) cols per pass
    const int c4 = tid & 31;
    const int r0 = tid >> 5;

    v8f acc[4][4];                        // 64x64 wave tile = 4x4 WMMA tiles
#pragma unroll
    for (int mt = 0; mt < 4; ++mt)
#pragma unroll
        for (int nt = 0; nt < 4; ++nt)
#pragma unroll
            for (int i = 0; i < 8; ++i) acc[mt][nt][i] = 0.0f;

    for (int kt = 0; kt < NUM_GROUPS; ++kt) {
        const int kBase = kt * BK;
        // branch-free prefetch target: next k-tile (clamped; re-prefetching the
        // last tile is harmless and keeps the staging loops control-flow free)
        const int kPref = (kt + 1 < NUM_GROUPS) ? (kBase + BK) : kBase;

        // ---- stage A: x[128][128] f32 -> f16 in LDS ----
        // unroll capped at 4 to bound live global-load registers (< 256 VGPRs,
        // avoids s_set_vgpr_msb traffic and keeps 2 blocks per WGP resident)
#pragma unroll 4
        for (int it = 0; it < BM / 8; ++it) {
            const int r = it * 8 + r0;
            const float* rowbase = x + (size_t)(mBase + r) * IN_F;
            const float4 v = *((const float4*)(rowbase + kBase) + c4);
            __builtin_prefetch((const void*)((const float4*)(rowbase + kPref) + c4), 0, 1);
            _Float16* dst = &lA[r * LSTR + c4 * 4];
            dst[0] = (_Float16)v.x; dst[1] = (_Float16)v.y;
            dst[2] = (_Float16)v.z; dst[3] = (_Float16)v.w;
        }

        // ---- stage B: dequant weight_q[256][128] int32 * scale -> f16 in LDS ----
#pragma unroll 4
        for (int it = 0; it < BN / 8; ++it) {
            const int r  = it * 8 + r0;
            const int gn = nBase + r;
            const float s = (float)scales[(size_t)gn * NUM_GROUPS + kt];
            const int* rowbase = wq + (size_t)gn * IN_F;
            const int4 q = *((const int4*)(rowbase + kBase) + c4);
            __builtin_prefetch((const void*)((const int4*)(rowbase + kPref) + c4), 0, 1);
            _Float16* dst = &lB[r * LSTR + c4 * 4];
            dst[0] = (_Float16)((float)q.x * s);
            dst[1] = (_Float16)((float)q.y * s);
            dst[2] = (_Float16)((float)q.z * s);
            dst[3] = (_Float16)((float)q.w * s);
        }

        __syncthreads();

        // ---- compute: 4 K-steps of 32, 16 WMMA per step per wave ----
#pragma unroll
        for (int kk = 0; kk < BK; kk += 32) {
            // A fragment (16-bit A 16x32 layout):
            // lanes 0-15: K=[0..7] then [16..23] ; lanes 16-31: K=[8..15] then [24..31]
            const int k0 = (lane < 16) ? 0 : 8;
            v16h afrag[4];
#pragma unroll
            for (int mt = 0; mt < 4; ++mt) {
                const _Float16* rowp =
                    &lA[(wm * 64 + mt * 16 + (lane & 15)) * LSTR + kk];
                v8h lo = *(const v8h*)(rowp + k0);
                v8h hi = *(const v8h*)(rowp + 16 + k0);
                afrag[mt] = cat8(lo, hi);
            }

            // B fragment (16-bit B 32x16 layout): lane col = lane&15,
            // lanes 0-15: K=[0..7] then [16..23] ; lanes 16-31: K=[8..15] then [24..31]
            const int kb = kk + ((lane < 16) ? 0 : 8);
            v16h bfrag[4];
#pragma unroll
            for (int nt = 0; nt < 4; ++nt) {
                const _Float16* rowp =
                    &lB[(wn * 64 + nt * 16 + (lane & 15)) * LSTR + kb];
                v8h lo = *(const v8h*)rowp;
                v8h hi = *(const v8h*)(rowp + 16);
                bfrag[nt] = cat8(lo, hi);
            }

#pragma unroll
            for (int mt = 0; mt < 4; ++mt)
#pragma unroll
                for (int nt = 0; nt < 4; ++nt)
                    acc[mt][nt] = __builtin_amdgcn_wmma_f32_16x16x32_f16(
                        false, afrag[mt], false, bfrag[nt],
                        (short)0, acc[mt][nt], false, false);
        }

        __syncthreads();
    }

    // ---- epilogue: bias add + store (16x16 f32 D layout) ----
    const int colInTile = lane & 15;
    const int rowHalf   = (lane < 16) ? 0 : 8;
#pragma unroll
    for (int nt = 0; nt < 4; ++nt) {
        const int gcol = nBase + wn * 64 + nt * 16 + colInTile;
        const float b  = bias[gcol];
#pragma unroll
        for (int mt = 0; mt < 4; ++mt) {
#pragma unroll
            for (int v = 0; v < 8; ++v) {
                const int grow = mBase + wm * 64 + mt * 16 + rowHalf + v;
                out[(size_t)grow * OUT_F + gcol] = acc[mt][nt][v] + b;
            }
        }
    }
}

extern "C" void kernel_launch(void* const* d_in, const int* in_sizes, int n_in,
                              void* d_out, int out_size, void* d_ws, size_t ws_size,
                              hipStream_t stream) {
    const float*    x     = (const float*)d_in[0];
    const int*      wq    = (const int*)d_in[1];
    const _Float16* sc    = (const _Float16*)d_in[2];
    const float*    bias  = (const float*)d_in[3];
    float*          out   = (float*)d_out;

    dim3 grid(M_TOTAL / BM, OUT_F / BN);   // 64 (M, fastest) x 43 (N)
    QuantLinear_72189810311631_kernel<<<grid, 256, 0, stream>>>(x, wq, sc, bias, out);
}
